// LocalAggregation_88235808129098
// MI455X (gfx1250) — matrix operand order repaired and checked by reading, not compiled, not run on hardware
//
#include <hip/hip_runtime.h>

#define BB 2
#define CC 64
#define NN 8192
#define KK 32
#define RAD 0.1f
#define R2 0.01f
#define BN_EPS 1e-5f

typedef __attribute__((ext_vector_type(16))) _Float16 v16h;
typedef __attribute__((ext_vector_type(8)))  float    v8f;

union HF16x16 { uint4 u[2]; v16h h; };

static __device__ __forceinline__ unsigned pkh(float a, float b) {
  union { _Float16 h[2]; unsigned u; } x;
  x.h[0] = (_Float16)a; x.h[1] = (_Float16)b;
  return x.u;
}

// ---------------------------------------------------------------------------
// Prep W into the CDNA5 WMMA B-matrix (32x16 f16) lane layout, 3 k-steps x 4
// n-tiles, stored as [s][t][lane][vgpr] dwords so each lane loads 2x b128.
// B layout: lane<16 -> N=lane, K={0..7,16..23} in V0..V7 (pairs); lane>=16 -> K={8..15,24..31}.
// ---------------------------------------------------------------------------
__global__ void prep_w_kernel(const float* __restrict__ W, unsigned* __restrict__ Wb) {
  int e = blockIdx.x * blockDim.x + threadIdx.x;   // 0..3071
  if (e >= 3072) return;
  int r    = e & 7;
  int lane = (e >> 3) & 31;
  int t    = (e >> 8) & 3;
  int s    = e >> 10;
  int ch = t * 16 + (lane & 15);                   // output channel (B's N dim)
  int k  = s * 32 + ((lane & 16) ? 8 : 0) + 2 * r + ((r >= 4) ? 8 : 0);
  float w0 = (k     < 67) ? W[ch * 67 + k    ] : 0.f;
  float w1 = (k + 1 < 67) ? W[ch * 67 + k + 1] : 0.f;
  Wb[e] = pkh(w0, w1);
}

// ---------------------------------------------------------------------------
// Transpose features (B,C,N) f32 -> (B,N,C) f16 so a neighbor's 64 channels
// are one contiguous 128B row (gather-friendly, L2 resident).
// ---------------------------------------------------------------------------
__global__ void prep_fea_kernel(const float* __restrict__ fea, _Float16* __restrict__ feaT) {
  int b = blockIdx.x >> 5;
  int n = (blockIdx.x & 31) * 256 + (int)threadIdx.x;
  uint4* dst = reinterpret_cast<uint4*>(feaT + ((size_t)(b * NN + n)) * 64);
#pragma unroll
  for (int p = 0; p < 8; ++p) {
    float f[8];
#pragma unroll
    for (int j = 0; j < 8; ++j) f[j] = fea[(size_t)(b * 64 + (p * 8 + j)) * NN + n];
    uint4 v;
    v.x = pkh(f[0], f[1]); v.y = pkh(f[2], f[3]);
    v.z = pkh(f[4], f[5]); v.w = pkh(f[6], f[7]);
    dst[p] = v;
  }
}

// ---------------------------------------------------------------------------
// Brute-force kNN (K=32). One thread per query, coords streamed via LDS in
// 2048-point chunks. Register-resident sorted top-32, branchless unrolled
// insert (static indices -> pure VGPRs). Radius rule applied at writeout.
// ---------------------------------------------------------------------------
__global__ void knn_kernel(const float* __restrict__ coor, int* __restrict__ idxb) {
  __shared__ float sx[2048], sy[2048], sz[2048];
  int b = blockIdx.x >> 5;
  int i = (blockIdx.x & 31) * 256 + (int)threadIdx.x;
  const float* cb = coor + (size_t)b * 3 * NN;
  float qx = cb[i], qy = cb[NN + i], qz = cb[2 * NN + i];

  float bd[32]; int bi[32];
#pragma unroll
  for (int s = 0; s < 32; ++s) { bd[s] = 1e30f; bi[s] = 0; }

  for (int ch = 0; ch < 4; ++ch) {
    int base = ch * 2048;
    __syncthreads();
#pragma unroll
    for (int r = 0; r < 8; ++r) {
      int j = r * 256 + (int)threadIdx.x;
      sx[j] = cb[base + j];
      sy[j] = cb[NN + base + j];
      sz[j] = cb[2 * NN + base + j];
    }
    __syncthreads();
#pragma unroll 4
    for (int j = 0; j < 2048; ++j) {
      float dx = qx - sx[j], dy = qy - sy[j], dz = qz - sz[j];
      float d = fmaf(dx, dx, fmaf(dy, dy, dz * dz));
      if (d < bd[31]) {
        float cd = d; int ci = base + j;
#pragma unroll
        for (int s = 0; s < 32; ++s) {
          bool sw = cd < bd[s];
          float td = bd[s]; int ti = bi[s];
          bd[s] = sw ? cd : td; bi[s] = sw ? ci : ti;
          cd    = sw ? td : cd; ci    = sw ? ti : ci;
        }
      }
    }
  }
  int nearest = bi[0];
  int* op = idxb + ((size_t)(b * NN + i)) * 32;
#pragma unroll
  for (int s = 0; s < 32; ++s) op[s] = (bd[s] > R2) ? nearest : bi[s];
}

// ---------------------------------------------------------------------------
// Fused gather + WMMA GEMM + BN-stat sums + per-query channel max/min.
// Block = 256 thr = 8 waves = 4 queries x 2 half-tiles (16 neighbors each).
// Each wave: stage its 16x96 (padded) g-tile in LDS as f16, then 3 k-steps
// x 4 n-tiles of v_wmma_f32_16x16x32_f16 against register-held W fragments.
// Epilogue folds max/min/sum/sumsq per channel; block writes deterministic
// partials (no float atomics).
// ---------------------------------------------------------------------------
__global__ void __launch_bounds__(256) gemm_kernel(
    const float* __restrict__ coor, const _Float16* __restrict__ feaT,
    const int* __restrict__ idxb, const unsigned* __restrict__ Wb,
    float* __restrict__ maxh, float* __restrict__ minh, float* __restrict__ partial) {
  __shared__ _Float16 slab[8][16 * 104];   // stride 104 halves = 208B (16B aligned, conflict-free)
  __shared__ float red[8][64][4];          // [wave][channel][max,min,sum,sumsq]

  int wave = (int)threadIdx.x >> 5;
  int lane = (int)threadIdx.x & 31;
  int b     = blockIdx.x / (NN / 4);
  int qbase = (blockIdx.x % (NN / 4)) * 4;
  int q     = qbase + (wave >> 1);
  int hsel  = wave & 1;
  const int* idxw = idxb + ((size_t)(b * NN + q)) * 32 + hsel * 16;

  // Gather 64 feature halves per neighbor row (2 lanes per row, 64B each).
  {
    int r  = lane >> 1;
    int nb = idxw[r];
    const uint4* src = reinterpret_cast<const uint4*>(
        feaT + ((size_t)(b * NN + nb)) * 64 + (lane & 1) * 32);
    uint4* dst = reinterpret_cast<uint4*>(&slab[wave][r * 104 + (lane & 1) * 32]);
#pragma unroll
    for (int p = 0; p < 4; ++p) dst[p] = src[p];
  }
  // Relative coords / RADIUS into k=64..66, zeros through k=95 (3rd k-step pad).
  if (lane < 16) {
    int r  = lane;
    int nb = idxw[r];
    const float* cb = coor + (size_t)b * 3 * NN;
    float cx = (cb[nb]          - cb[q])          * (1.f / RAD);
    float cy = (cb[NN + nb]     - cb[NN + q])     * (1.f / RAD);
    float cz = (cb[2 * NN + nb] - cb[2 * NN + q]) * (1.f / RAD);
    uint4 v0; v0.x = pkh(cx, cy); v0.y = pkh(cz, 0.f); v0.z = 0u; v0.w = 0u;
    uint4 vz; vz.x = 0u; vz.y = 0u; vz.z = 0u; vz.w = 0u;
    uint4* dst = reinterpret_cast<uint4*>(&slab[wave][r * 104 + 64]);
    dst[0] = v0; dst[1] = vz; dst[2] = vz; dst[3] = vz;
  }
  __syncthreads();

  // W fragments (pre-swizzled): 12 x v16h, two b128 loads each.
  HF16x16 wf[3][4];
  const uint4* wbp = reinterpret_cast<const uint4*>(Wb);
#pragma unroll
  for (int s = 0; s < 3; ++s)
#pragma unroll
    for (int t = 0; t < 4; ++t) {
      int base = ((s * 4 + t) * 32 + lane) * 2;
      wf[s][t].u[0] = wbp[base];
      wf[s][t].u[1] = wbp[base + 1];
    }

  v8f acc[4];
  v8f vzero = {0.f, 0.f, 0.f, 0.f, 0.f, 0.f, 0.f, 0.f};
#pragma unroll
  for (int t = 0; t < 4; ++t) acc[t] = vzero;

  // A fragment per CDNA5 16-bit A layout: lane<16 takes K {0..7,16..23},
  // lane>=16 takes K {8..15,24..31} of its row m = lane&15.
  int m  = lane & 15;
  int hw = (lane >> 4) << 3;  // half-offset 0 or 8
#pragma unroll
  for (int s = 0; s < 3; ++s) {
    HF16x16 af;
    const _Float16* row = &slab[wave][m * 104 + s * 32];
    af.u[0] = *reinterpret_cast<const uint4*>(row + hw);
    af.u[1] = *reinterpret_cast<const uint4*>(row + 16 + hw);
#pragma unroll
    for (int t = 0; t < 4; ++t)
      acc[t] = __builtin_amdgcn_wmma_f32_16x16x32_f16(
          false, af.h, false, wf[s][t].h, (short)0, acc[t], false, false);
  }

  // Epilogue: per-channel max/min/sum/sumsq over the 16 rows (M dim).
  // D layout: VGPR r -> M=r (lanes 0-15) / M=8+r (lanes 16-31), N = lane&15.
#pragma unroll
  for (int t = 0; t < 4; ++t) {
    float mx = -1e30f, mn = 1e30f, sm = 0.f, sq = 0.f;
#pragma unroll
    for (int r = 0; r < 8; ++r) {
      float e = acc[t][r];
      mx = fmaxf(mx, e); mn = fminf(mn, e);
      sm += e; sq = fmaf(e, e, sq);
    }
    mx = fmaxf(mx, __shfl_xor(mx, 16));
    mn = fminf(mn, __shfl_xor(mn, 16));
    sm += __shfl_xor(sm, 16);
    sq += __shfl_xor(sq, 16);
    if (lane < 16) {
      int c = t * 16 + lane;
      red[wave][c][0] = mx; red[wave][c][1] = mn;
      red[wave][c][2] = sm; red[wave][c][3] = sq;
    }
  }
  __syncthreads();

  if ((wave & 1) == 0) {            // combine the two half-tiles of each query
    int qq = qbase + (wave >> 1);
#pragma unroll
    for (int j = 0; j < 2; ++j) {
      int c = lane + j * 32;
      float mx = fmaxf(red[wave][c][0], red[wave + 1][c][0]);
      float mn = fminf(red[wave][c][1], red[wave + 1][c][1]);
      size_t o = ((size_t)(b * NN + qq)) * 64 + c;
      maxh[o] = mx; minh[o] = mn;
    }
  }
  if (wave == 0) {                  // deterministic per-block BN partials
#pragma unroll
    for (int j = 0; j < 2; ++j) {
      int c = lane + j * 32;
      float S = 0.f, Q = 0.f;
#pragma unroll
      for (int w = 0; w < 8; ++w) { S += red[w][c][2]; Q += red[w][c][3]; }
      partial[(size_t)blockIdx.x * 128 + c]      = S;
      partial[(size_t)blockIdx.x * 128 + 64 + c] = Q;
    }
  }
}

// ---------------------------------------------------------------------------
// Reduce 4096 block partials -> per-channel mean and gamma*rsqrt(var+eps).
// (Conv bias cancels exactly under training-mode BN, so it is omitted.)
// ---------------------------------------------------------------------------
__global__ void stats_kernel(const float* __restrict__ partial,
                             const float* __restrict__ gamma,
                             float* __restrict__ ms) {
  __shared__ float ls[4][64], lq[4][64];
  int c = (int)threadIdx.x & 63, sl = (int)threadIdx.x >> 6;
  float S = 0.f, Q = 0.f;
  for (int blk = sl; blk < 4096; blk += 4) {
    S += partial[(size_t)blk * 128 + c];
    Q += partial[(size_t)blk * 128 + 64 + c];
  }
  ls[sl][c] = S; lq[sl][c] = Q;
  __syncthreads();
  if (threadIdx.x < 64) {
    float s  = ls[0][c] + ls[1][c] + ls[2][c] + ls[3][c];
    float qv = lq[0][c] + lq[1][c] + lq[2][c] + lq[3][c];
    const float inv = 1.f / (float)((size_t)BB * NN * KK);
    float mean  = s * inv;
    float var   = qv * inv - mean * mean;
    float scale = gamma[c] * rsqrtf(var + BN_EPS);
    ms[c * 2] = mean; ms[c * 2 + 1] = scale;
  }
}

// ---------------------------------------------------------------------------
// BN affine + ReLU applied to the commuted K-extreme; writes (B,C,N).
// ---------------------------------------------------------------------------
__global__ void final_kernel(const float* __restrict__ maxh, const float* __restrict__ minh,
                             const float* __restrict__ ms, const float* __restrict__ beta,
                             float* __restrict__ out) {
  int b = blockIdx.x >> 6;
  int c = blockIdx.x & 63;
  float mean = ms[c * 2], scale = ms[c * 2 + 1], bet = beta[c];
  const float* src = (scale >= 0.f) ? maxh : minh;
  for (int n = (int)threadIdx.x; n < NN; n += 256) {
    float x = src[((size_t)(b * NN + n)) * 64 + c];
    float v = fmaf(x - mean, scale, bet);
    out[((size_t)(b * 64 + c)) * NN + n] = fmaxf(v, 0.f);
  }
}

extern "C" void kernel_launch(void* const* d_in, const int* in_sizes, int n_in,
                              void* d_out, int out_size, void* d_ws, size_t ws_size,
                              hipStream_t stream) {
  const float* coor  = (const float*)d_in[0];  // (B,3,N)
  const float* fea   = (const float*)d_in[1];  // (B,C,N)
  const float* W     = (const float*)d_in[2];  // (C, C+3)
  // d_in[3] = conv bias: cancels exactly under training-mode BatchNorm.
  const float* gamma = (const float*)d_in[4];
  const float* beta  = (const float*)d_in[5];
  float* out = (float*)d_out;

  char* ws = (char*)d_ws;
  size_t off = 0;
  auto alloc = [&](size_t bytes) -> void* {
    void* p = ws + off;
    off = (off + bytes + 255) & ~(size_t)255;
    return p;
  };
  _Float16* feaT = (_Float16*)alloc((size_t)BB * NN * 64 * 2);  // 2 MB
  unsigned* Wb   = (unsigned*)alloc(3072 * 4);                  // 12 KB
  int*      idxb = (int*)     alloc((size_t)BB * NN * 32 * 4);  // 2 MB
  float*    maxh = (float*)   alloc((size_t)BB * NN * 64 * 4);  // 4 MB
  float*    minh = (float*)   alloc((size_t)BB * NN * 64 * 4);  // 4 MB
  float*    part = (float*)   alloc((size_t)4096 * 128 * 4);    // 2 MB
  float*    ms   = (float*)   alloc(64 * 2 * 4);

  prep_w_kernel  <<<12,   256, 0, stream>>>(W, Wb);
  prep_fea_kernel<<<64,   256, 0, stream>>>(fea, feaT);
  knn_kernel     <<<64,   256, 0, stream>>>(coor, idxb);
  gemm_kernel    <<<4096, 256, 0, stream>>>(coor, feaT, idxb, Wb, maxh, minh, part);
  stats_kernel   <<<1,    256, 0, stream>>>(part, gamma, ms);
  final_kernel   <<<128,  256, 0, stream>>>(maxh, minh, ms, beta, out);
}
